// MixedOp_51290499449114
// MI455X (gfx1250) — matrix operand order repaired and checked by reading, not compiled, not run on hardware
//
#include <hip/hip_runtime.h>
#include <hip/hip_bf16.h>

typedef __attribute__((ext_vector_type(16))) __bf16 v16bf;
typedef __attribute__((ext_vector_type(8)))  float  v8f;

#define NB 16
#define NC 256
#define CK 64
#define HH 64
#define WW 64
#define HW 4096
#define NOPS 6
#define KSTEPS 86   // conv3: 18, conv5: 50, convd: 18 (tap-major, 2 half-K chunks of 32 ci)
#define PW 68       // padded plane width/height (halo 2 each side)

#if defined(__AMDGCN__) && __has_builtin(__builtin_amdgcn_tensor_load_to_lds) && __has_builtin(__builtin_amdgcn_s_wait_tensorcnt)
#define USE_TDM 1
#else
#define USE_TDM 0
#endif

#if USE_TDM
typedef __attribute__((ext_vector_type(4))) unsigned tdm_v4u;
typedef __attribute__((ext_vector_type(8))) int      tdm_v8i;
typedef __attribute__((ext_vector_type(4))) int      tdm_v4i;

// 2-D TDM descriptor: tile = nrows rows of row8 8-byte units, global row stride = stride8
// units, destination rows contiguous in LDS.  D# layout per CDNA5 ISA §8 (group0/group1).
__device__ __forceinline__ void tdm_load_2d(unsigned lds_addr, const __bf16* gptr,
                                            unsigned row8, unsigned nrows, unsigned stride8) {
    unsigned long long ga = (unsigned long long)(uintptr_t)gptr;
    tdm_v4u g0;
    g0[0] = 1u;                                              // count=1, user descriptor
    g0[1] = lds_addr;                                        // LDS byte address
    g0[2] = (unsigned)(ga & 0xFFFFFFFFu);                    // global_addr[31:0]
    g0[3] = (unsigned)((ga >> 32) & 0x01FFFFFFu) | (2u << 30); // global_addr[56:32], type=2
    tdm_v8i g1 = {0, 0, 0, 0, 0, 0, 0, 0};
    g1[0] = (int)(3u << 16);                                 // data_size=8B, mask=0, no flags
    g1[1] = (int)((stride8 & 0xFFFFu) << 16);                // tensor_dim0 lo16 (= stride)
    g1[2] = (int)(0xFFFFu << 16);                            // tensor_dim0 hi=0, tensor_dim1 lo=big
    g1[3] = (int)((row8 & 0xFFFFu) << 16);                   // tensor_dim1 hi=0, tile_dim0
    g1[4] = (int)(nrows & 0xFFFFu);                          // tile_dim1, tile_dim2=0
    g1[5] = (int)stride8;                                    // tensor_dim0_stride lo32
    tdm_v4i z4 = {0, 0, 0, 0};
#if defined(__clang_major__) && __clang_major__ >= 23
    tdm_v8i z8 = {0, 0, 0, 0, 0, 0, 0, 0};
    __builtin_amdgcn_tensor_load_to_lds(g0, g1, z4, z4, z8, 0);
#else
    __builtin_amdgcn_tensor_load_to_lds(g0, g1, z4, z4, 0);
#endif
}
#endif

// ---------------- stage 1: per-(b,c) spatial mean & max ----------------
__global__ void k_avgmax(const float* __restrict__ x, float* avg, float* mxo) {
    __shared__ float ss[256], sm[256];
    int bc = blockIdx.x, tid = threadIdx.x;
    const float* p = x + (size_t)bc * HW;
    float s = 0.f, m = -3.0e38f;
    for (int i = tid; i < HW; i += 256) { float v = p[i]; s += v; m = fmaxf(m, v); }
    ss[tid] = s; sm[tid] = m; __syncthreads();
    for (int st = 128; st > 0; st >>= 1) {
        if (tid < st) { ss[tid] += ss[tid + st]; sm[tid] = fmaxf(sm[tid], sm[tid + st]); }
        __syncthreads();
    }
    if (tid == 0) { avg[bc] = ss[0] * (1.f / HW); mxo[bc] = sm[0]; }
}

// ---------------- stage 2: channel-attention MLP -> num_list ----------------
__global__ void k_mlp(const float* __restrict__ avg, const float* __restrict__ mx,
                      const float* __restrict__ w1, const float* __restrict__ w2,
                      float* num) {
    __shared__ float h[2][NB][128];
    int tid = threadIdx.x;
    for (int t = tid; t < 4096; t += 256) {
        int set = t >> 11, b = (t >> 7) & 15, j = t & 127;
        const float* v = set ? mx : avg;
        float s = 0.f;
        for (int c = 0; c < NC; c++) s += v[b * NC + c] * w1[j * NC + c];
        h[set][b][j] = fmaxf(s, 0.f);
    }
    __syncthreads();
    for (int t = tid; t < NB * NC; t += 256) {
        int b = t >> 8, c = t & 255;
        float s = 0.f;
        for (int j = 0; j < 128; j++) s += (h[0][b][j] + h[1][b][j]) * w2[c * 128 + j];
        num[t] = 1.f / (1.f + __expf(-s));
    }
}

// ---------------- stage 3: top-k(64) channel selection ----------------
__global__ void k_topk(const float* __restrict__ num, int* idx, int* chp) {
    __shared__ float s[NC];
    int c = threadIdx.x;
    float t = 0.f;
    for (int b = 0; b < NB; b++) t += num[b * NC + c];
    s[c] = t; __syncthreads();
    int rank = 0;
    for (int c2 = 0; c2 < NC; c2++) {
        float v = s[c2];
        if (v > t || (v == t && c2 < c)) rank++;
    }
    if (rank < CK) { idx[rank] = c; chp[c] = rank; } else chp[c] = -1;
}

// ---------------- stage 4: x2 = x*num (to d_out), gather xtemp(f32) ----------------
__global__ void k_scale(const float* __restrict__ x, const float* __restrict__ num,
                        const int* __restrict__ chp, float* __restrict__ out,
                        float* __restrict__ xtemp) {
    size_t i = (size_t)blockIdx.x * 256 + threadIdx.x;
    int c = (int)((i >> 12) & 255);
    int b = (int)(i >> 20);
    int p = (int)(i & 4095);
    float v = x[i] * num[b * NC + c];
    out[i] = v;
    int pos = chp[c];
    if (pos >= 0) xtemp[(((size_t)b * CK + pos) << 12) + p] = v;
}

// ---------------- stage 4b: zero padded channel-last arrays ----------------
__global__ void k_zpad(unsigned* __restrict__ p, int ndw) {
    int i = blockIdx.x * 256 + threadIdx.x;
    if (i < ndw) p[i] = 0u;
}

// ---------------- stage 4c: transpose to channel-last, halo-padded bf16 (raw + relu) ----
// xtr layout: [b][hp(68)][wp(68)][ci(64)] bf16, interior at hp=h+2, wp=w+2.
__global__ void k_transpose(const float* __restrict__ xtemp,
                            __bf16* __restrict__ xr, __bf16* __restrict__ xl) {
    __shared__ float t[WW][CK + 1];
    int bh = blockIdx.x;
    int b = bh >> 6, h = bh & 63;
    int tid = threadIdx.x;
    for (int i = tid; i < CK * WW; i += 256) {           // coalesced read: w fastest
        int ci = i >> 6, w = i & 63;
        t[w][ci] = xtemp[(((size_t)b * CK + ci) << 12) + (h << 6) + w];
    }
    __syncthreads();
    for (int i = tid; i < CK * WW; i += 256) {           // coalesced write: ci fastest
        int w = i >> 6, ci = i & 63;
        float v = t[w][ci];
        size_t o = (((size_t)b * PW + (h + 2)) * PW + (w + 2)) * CK + ci;
        xr[o] = (__bf16)v;
        xl[o] = (__bf16)fmaxf(v, 0.f);
    }
}

// ---------------- stage 5: pack conv weights into bf16 A-fragment layout ----------------
// A-fragment 16x32 bf16 as packed here (contiguous k, same order as B): lane holds row M=lane%16; its 16 values are
// K = g*16 + 0..15, g = lane/16.  One lane-fragment = 16 contiguous bf16.
__global__ void k_prepack(const float* __restrict__ w3, const float* __restrict__ w5,
                          const float* __restrict__ wd, __bf16* __restrict__ Wpk) {
    int t = blockIdx.x * 256 + threadIdx.x;
    if (t >= KSTEPS * 4 * 32) return;
    int lane = t & 31, rest = t >> 5;
    int kglob = rest % KSTEPS, cot = rest / KSTEPS;
    int s, kdim; const float* w;
    if (kglob < 18)      { s = kglob;      kdim = 3; w = w3; }
    else if (kglob < 68) { s = kglob - 18; kdim = 5; w = w5; }
    else                 { s = kglob - 68; kdim = 3; w = wd; }
    int tap = s >> 1, hk = s & 1;
    int kh = tap / kdim, kw = tap % kdim;
    int m = lane & 15, g = lane >> 4;
    int co = cot * 16 + m;
    __bf16* dst = Wpk + (((size_t)(kglob * 4 + cot) * 32 + lane) << 4);
    #pragma unroll
    for (int j = 0; j < 16; j++) {
        int k32 = g * 16 + j;
        int ci = hk * 32 + k32;
        dst[j] = (__bf16)w[(((size_t)co * CK + ci) * kdim + kh) * kdim + kw];
    }
}

// ---------------- stage 6: fused 6-op kernel (two passes) ----------------
// Block: 256 thr = 8 waves; handles (b, row h, 32 columns). Wave w: co-tile = w%4, pos-tile = w/4.
// Conv = implicit GEMM via v_wmma_f32_16x16x32_bf16 (M=co16, N=pos16, K=ci,tap).
// Tiles staged via Tensor Data Mover (2 x tensor_load_to_lds, 2-D descriptors).
template <bool STATS>
__global__ void k_fused(const __bf16* __restrict__ xtr_raw, const __bf16* __restrict__ xtr_rel,
                        const float* __restrict__ xtemp, const __bf16* __restrict__ Wpk,
                        float* S1, float* S2, float* Msp,
                        const float* __restrict__ A6, const float* __restrict__ Cs,
                        const int* __restrict__ idx, float* __restrict__ out) {
    __shared__ __align__(32) __bf16 tile_raw[5][36][CK];   // pre-ReLU (pools)
    __shared__ __align__(32) __bf16 tile_rel[5][36][CK];   // ReLU'd (conv B fragments)
    __shared__ float lA[NOPS * CK];
    __shared__ float lC[CK];
    __shared__ int   lidx[CK];

    int tid = threadIdx.x, blk = blockIdx.x;
    int b = blk >> 7, rem = blk & 127;
    int h = rem >> 1, posbase = (rem & 1) * 32;

    int lane = tid & 31, wv = tid >> 5;
    int cot = wv & 3, pt = wv >> 2;
    int nloc = lane & 15, g = lane >> 4;
    int xb = pt * 16;

    // tile origin in padded coords: row (h-2)+2 = h, col (posbase-2)+2 = posbase
    size_t gbase = (((size_t)b * PW + h) * PW + posbase) * CK;

#if USE_TDM
    if (wv == 0) {
        // 5 rows x (36*64*2 = 4608B = 576 u8) with stride 68*64*2 = 8704B = 1088 u8
        tdm_load_2d((unsigned)(uintptr_t)&tile_raw[0][0][0], xtr_raw + gbase, 576u, 5u, 1088u);
        tdm_load_2d((unsigned)(uintptr_t)&tile_rel[0][0][0], xtr_rel + gbase, 576u, 5u, 1088u);
        __builtin_amdgcn_s_wait_tensorcnt(0);
    }
#else
    for (int t = tid; t < 5 * 36 * CK; t += 256) {
        int ci = t & 63, xx = (t >> 6) % 36, r = t / (36 * CK);
        size_t o = gbase + ((size_t)r * PW + xx) * CK + ci;
        tile_raw[r][xx][ci] = xtr_raw[o];
        tile_rel[r][xx][ci] = xtr_rel[o];
    }
#endif
    if (!STATS) {
        for (int t = tid; t < NOPS * CK; t += 256) lA[t] = A6[b * NOPS * CK + t];
        for (int t = tid; t < CK; t += 256)        lC[t] = Cs[b * CK + t];
        for (int t = tid; t < CK; t += 256)        lidx[t] = idx[t];
    }
    __syncthreads();

    __builtin_prefetch(Wpk + ((size_t)lane << 4), 0, 1);

    // B-fragment 32x16 bf16: lane holds column N=lane%16, K = (lane/16)*16 + 0..15
    // -> 16 contiguous channels in LDS (already ReLU'd), pure ds_load, no VALU.
    const __bf16* bbase = &tile_rel[2][xb + nloc + 2][g * 16];
    auto buildB = [&](int oy, int ox, int hk) -> v16bf {
        return *(const v16bf*)(bbase + ((oy * 36 + ox) * CK + hk * 32));
    };
    auto loadA = [&](int kglob) -> v16bf {
        return *(const v16bf*)(Wpk + (((size_t)(kglob * 4 + cot) * 32 + lane) << 4));
    };

    v8f a3 = {0,0,0,0,0,0,0,0}, a5 = {0,0,0,0,0,0,0,0}, ad = {0,0,0,0,0,0,0,0};

    int kg = 0;
    for (int ky = -1; ky <= 1; ky++)                 // 3x3, dil 1
        for (int kx = -1; kx <= 1; kx++) {
            #pragma unroll
            for (int hk = 0; hk < 2; hk++) {
                v16bf av = loadA(kg + hk);
                v16bf bv = buildB(ky, kx, hk);
                a3 = __builtin_amdgcn_wmma_f32_16x16x32_bf16(false, av, false, bv, (short)0, a3, false, false);
            }
            kg += 2;
        }
    for (int ky = -2; ky <= 2; ky++)                 // 5x5, dil 1
        for (int kx = -2; kx <= 2; kx++) {
            #pragma unroll
            for (int hk = 0; hk < 2; hk++) {
                v16bf av = loadA(kg + hk);
                v16bf bv = buildB(ky, kx, hk);
                a5 = __builtin_amdgcn_wmma_f32_16x16x32_bf16(false, av, false, bv, (short)0, a5, false, false);
            }
            kg += 2;
        }
    for (int ky = -2; ky <= 2; ky += 2)              // 3x3, dil 2
        for (int kx = -2; kx <= 2; kx += 2) {
            #pragma unroll
            for (int hk = 0; hk < 2; hk++) {
                v16bf av = loadA(kg + hk);
                v16bf bv = buildB(ky, kx, hk);
                ad = __builtin_amdgcn_wmma_f32_16x16x32_bf16(false, av, false, bv, (short)0, ad, false, false);
            }
            kg += 2;
        }

    // Pools + identity. C/D layout: vgpr r -> M = r + 8*(lane/16); lane%16 -> N.
    int gw = posbase + xb + nloc;
    float idv[8], mpv[8], apv[8];
    #pragma unroll
    for (int r2 = 0; r2 < 8; r2++) {
        int m = r2 + 8 * g, co = cot * 16 + m;
        int xx = xb + nloc + 2;
        idv[r2] = xtemp[(((size_t)b * CK + co) << 12) + (h << 6) + gw];  // exact f32 identity
        float mxv = -3.0e38f, sm = 0.f;
        #pragma unroll
        for (int dy = -1; dy <= 1; dy++) {
            int gh = h + dy;
            #pragma unroll
            for (int dx = -1; dx <= 1; dx++) {
                float v = (float)tile_raw[2 + dy][xx + dx][co];
                sm += v;                                   // zero-padded => /9 matches ref
                int gw2 = gw + dx;
                if ((unsigned)gh < 64u && (unsigned)gw2 < 64u) mxv = fmaxf(mxv, v);
            }
        }
        mpv[r2] = mxv; apv[r2] = sm * (1.f / 9.f);
    }

    if (STATS) {
        auto redsum = [](float v) {
            #pragma unroll
            for (int m2 = 1; m2 < 16; m2 <<= 1) v += __shfl_xor(v, m2, 32);
            return v;
        };
        #pragma unroll
        for (int r2 = 0; r2 < 8; r2++) {
            int m = r2 + 8 * g, co = cot * 16 + m;
            float vv[NOPS] = { mpv[r2], apv[r2], idv[r2], a3[r2], a5[r2], ad[r2] };
            #pragma unroll
            for (int i = 0; i < NOPS; i++) {
                float s1 = redsum(vv[i]);
                float s2 = redsum(vv[i] * vv[i]);
                if (nloc == 0) {
                    atomicAdd(&S1[i * CK + co], s1);
                    atomicAdd(&S2[i * CK + co], s2);
                    atomicAdd(&Msp[(b * NOPS + i) * CK + co], s1);
                }
            }
        }
    } else {
        #pragma unroll
        for (int r2 = 0; r2 < 8; r2++) {
            int m = r2 + 8 * g, co = cot * 16 + m;
            float o = lA[0 * CK + co] * mpv[r2] + lA[1 * CK + co] * apv[r2]
                    + lA[2 * CK + co] * idv[r2] + lA[3 * CK + co] * a3[r2]
                    + lA[4 * CK + co] * a5[r2] + lA[5 * CK + co] * ad[r2]
                    - lC[co];
            int ch = lidx[co];
            out[(((size_t)b * NC + ch) << 12) + (h << 6) + gw] = o;
        }
    }
}

// ---------------- stage 7: bn stats -> attention MLP -> y, combine coeffs, op_attention ----
__global__ void k_att(const float* __restrict__ S1, const float* __restrict__ S2,
                      const float* __restrict__ Msp,
                      const float* __restrict__ w1, const float* __restrict__ w2,
                      float* A6, float* Cs, float* opatt) {
    __shared__ float m_s[384], inv_s[384];
    __shared__ float tm[NB][384];
    __shared__ float hid[NB][48];
    __shared__ float ys[NB][384];
    int tid = threadIdx.x;
    const float Ninv = 1.f / (NB * HW);
    for (int t = tid; t < 384; t += 256) {
        int i = t >> 6;
        if (i == 2) { m_s[t] = 0.f; inv_s[t] = 1.f; }            // identity group: no bn
        else {
            float mu = S1[t] * Ninv;
            float var = S2[t] * Ninv - mu * mu;
            m_s[t] = mu; inv_s[t] = rsqrtf(var + 1e-5f);
        }
    }
    __syncthreads();
    for (int t = tid; t < NB * 384; t += 256) {
        int j = t % 384;
        tm[t / 384][j] = (Msp[t] * (1.f / HW) - m_s[j]) * inv_s[j];
    }
    __syncthreads();
    for (int t = tid; t < NB * 48; t += 256) {
        int b = t / 48, k = t % 48;
        float s = 0.f;
        for (int j = 0; j < 384; j++) s += tm[b][j] * w1[k * 384 + j];
        hid[b][k] = fmaxf(s, 0.f);
    }
    __syncthreads();
    for (int t = tid; t < NB * 384; t += 256) {
        int b = t / 384, j = t % 384;
        float s = 0.f;
        for (int k = 0; k < 48; k++) s += hid[b][k] * w2[j * 48 + k];
        float y = 1.f / (1.f + __expf(-s));
        ys[b][j] = y;
        A6[t] = y * inv_s[j];
    }
    __syncthreads();
    for (int t = tid; t < NB * CK; t += 256) {
        int b = t >> 6, ck = t & 63;
        float c = 0.f;
        for (int i = 0; i < NOPS; i++) { int j = i * CK + ck; c += ys[b][j] * inv_s[j] * m_s[j]; }
        Cs[t] = c;
    }
    if (tid < NOPS) {
        float ss = 0.f;
        for (int b = 0; b < NB; b++) for (int ck = 0; ck < CK; ck++) ss += ys[b][tid * CK + ck];
        float mean = ss * (1.f / 1024.f);
        float var = 0.f;
        for (int b = 0; b < NB; b++) for (int ck = 0; ck < CK; ck++) {
            float d = ys[b][tid * CK + ck] - mean; var += d * d;
        }
        var *= (1.f / 1023.f);
        opatt[tid] = ss - sqrtf(var);
    }
}

__global__ void k_zero(float* p) {
    for (int t = threadIdx.x; t < 6912; t += 256) p[t] = 0.f;   // S1,S2,Msp contiguous
}

// ---------------- launch ----------------
extern "C" void kernel_launch(void* const* d_in, const int* in_sizes, int n_in,
                              void* d_out, int out_size, void* d_ws, size_t ws_size,
                              hipStream_t stream) {
    const float* x      = (const float*)d_in[0];
    const float* ca_w1  = (const float*)d_in[1];
    const float* ca_w2  = (const float*)d_in[2];
    const float* att_w1 = (const float*)d_in[3];
    const float* att_w2 = (const float*)d_in[4];
    const float* w3     = (const float*)d_in[5];
    const float* w5     = (const float*)d_in[6];
    const float* wd     = (const float*)d_in[7];
    float* out = (float*)d_out;

    float* f      = (float*)d_ws;
    float* avg    = f + 0;        // 4096
    float* mx     = f + 4096;     // 4096
    float* num    = f + 8192;     // 4096
    float* S1     = f + 12288;    // 384
    float* S2     = f + 12672;    // 384
    float* Msp    = f + 13056;    // 6144
    float* A6     = f + 19200;    // 6144
    float* Cs     = f + 25344;    // 1024
    int*   idx    = (int*)(f + 26368);   // 64
    int*   chp    = (int*)(f + 26432);   // 256
    __bf16* Wpk   = (__bf16*)(f + 26688);        // 176128 bf16 (32B aligned)
    float* xtemp  = f + 114752;   // 4194304 f32 gathered x2
    __bf16* xtrr  = (__bf16*)(f + 4309056);      // 16*68*68*64 bf16 raw, halo-padded
    __bf16* xtrl  = (__bf16*)(f + 6676544);      // same, ReLU'd
    const int npadded_dw = 2 * (NB * PW * PW * CK / 2);       // both arrays, as dwords

    k_zero<<<1, 256, 0, stream>>>(S1);
    k_avgmax<<<NB * NC, 256, 0, stream>>>(x, avg, mx);
    k_mlp<<<1, 256, 0, stream>>>(avg, mx, ca_w1, ca_w2, num);
    k_topk<<<1, NC, 0, stream>>>(num, idx, chp);
    k_scale<<<(NB * NC * HW) / 256, 256, 0, stream>>>(x, num, chp, out, xtemp);
    k_zpad<<<(npadded_dw + 255) / 256, 256, 0, stream>>>((unsigned*)xtrr, npadded_dw);
    k_transpose<<<NB * HH, 256, 0, stream>>>(xtemp, xtrr, xtrl);
    k_prepack<<<(KSTEPS * 4 * 32 + 255) / 256, 256, 0, stream>>>(w3, w5, wd, Wpk);
    k_fused<true><<<NB * HH * 2, 256, 0, stream>>>(xtrr, xtrl, xtemp, Wpk, S1, S2, Msp,
                                                   nullptr, nullptr, nullptr, nullptr);
    k_att<<<1, 256, 0, stream>>>(S1, S2, Msp, att_w1, att_w2, A6, Cs, out + (size_t)NB * NC * HW);
    k_fused<false><<<NB * HH * 2, 256, 0, stream>>>(xtrr, xtrl, xtemp, Wpk, S1, S2, Msp,
                                                    A6, Cs, idx, out);
    (void)in_sizes; (void)n_in; (void)out_size; (void)ws_size;
}
